// GCNResnet_7868380086329
// MI455X (gfx1250) — matrix operand order repaired and checked by reading, not compiled, not run on hardware
//
#include <hip/hip_runtime.h>

// ---------------------------------------------------------------------------
// GCNResnet forward for MI455X (gfx1250, wave32, WMMA bf16 16x16x32).
//
// Refactor: out = (feature@W_img^T + b_img) @ cls^T
//             == feature @ (cls @ W_img)^T + (b_img . cls^T)
// cuts FLOPs ~6x (37.6 -> 6.5 GFLOP) and removes the [1024,8000]
// intermediate; the whole net is then one memory-bound pass over
// W_img/W_cls (~145 MB -> ~6.3 us @ 23.3 TB/s HBM).
//
// GEMM kernel: 8 waves/block, block tile 16(M) x 128(N), K-step 64
// (2 x v_wmma_f32_16x16x32_bf16 per stage). Uniform fast paths do
// b128 global loads + packed b64/b128 LDS stores; edge/tail blocks use a
// branchless (clamped-index + select) slow path to avoid exec-mask churn.
// ---------------------------------------------------------------------------

typedef __attribute__((ext_vector_type(16))) __bf16    v16bf;
typedef __attribute__((ext_vector_type(8)))  float     v8f;
typedef __attribute__((ext_vector_type(4)))  float     fl4;
typedef __attribute__((ext_vector_type(2)))  unsigned  u32x2;

union FragBF16 {
    v16bf    v;
    unsigned u[8];
};

__device__ __forceinline__ unsigned short f32_to_bf16(float f) {
    union { float f; unsigned u; } x; x.f = f;
    unsigned r = x.u + 0x7FFFu + ((x.u >> 16) & 1u);   // round-to-nearest-even
    return (unsigned short)(r >> 16);
}
__device__ __forceinline__ float bf16_to_f32(unsigned short h) {
    union { unsigned u; float f; } x; x.u = ((unsigned)h) << 16;
    return x.f;
}
__device__ __forceinline__ unsigned pack2_bf16(float lo, float hi) {
    return ((unsigned)f32_to_bf16(hi) << 16) | (unsigned)f32_to_bf16(lo);
}

template <typename AT>
__device__ __forceinline__ unsigned short load_a_elem(const AT* p) {
    if constexpr (sizeof(AT) == 4) return f32_to_bf16(*(const float*)p);
    else                           return *(const unsigned short*)p;
}

// stage 4 consecutive A elements (along k) into LDS as 4 bf16 (8 bytes)
template <typename AT>
__device__ __forceinline__ void stage_a4(const AT* p, unsigned short* dst) {
    unsigned* d = (unsigned*)dst;
    if constexpr (sizeof(AT) == 4) {
        fl4 r = *(const fl4*)p;                 // global_load_b128
        d[0] = pack2_bf16(r[0], r[1]);
        d[1] = pack2_bf16(r[2], r[3]);          // ds_store_b64
    } else {
        u32x2 r = *(const u32x2*)p;             // global_load_b64 (bf16 src)
        d[0] = r[0];
        d[1] = r[1];
    }
}

// ---------------------------------------------------------------------------
// C[M,N] = act( A[M,K] @ B + bias[N] )
//   AT: float | unsigned short(bf16)   TRANS_B: B is [N,K] if true
//   ACT: 0 none, 1 leaky(0.2)          OT: float | unsigned short(bf16)
// Grid: (ceil(N/128), M/16), 256 threads. M must be a multiple of 16.
// ---------------------------------------------------------------------------
template <typename AT, bool TRANS_B, int ACT, typename OT>
__global__ __launch_bounds__(256)
void gemm_wmma_bf16(const AT* __restrict__ A, const float* __restrict__ B,
                    const float* __restrict__ bias, OT* __restrict__ C,
                    int M, int N, int K) {
    __shared__ unsigned short sA[16 * 64];      // [m][k]
    __shared__ unsigned short sB[128 * 64];     // [n_local][k] (k contiguous)

    const int t    = threadIdx.x;
    const int lane = t & 31;
    const int wave = t >> 5;
    const int half = lane >> 4;
    const int lr   = lane & 15;

    const int  n0    = blockIdx.x * 128;
    const int  m0    = blockIdx.y * 16;
    const bool nFull = (n0 + 128 <= N);         // block-uniform

    v8f acc = {};

    for (int k0 = 0; k0 < K; k0 += 64) {
        const bool kFull = (k0 + 64 <= K);      // block-uniform

        // ================= stage A tile [16 x 64] =================
        if (kFull) {
            const int row = t >> 4;
            const int k4  = (t & 15) * 4;
            stage_a4(A + (long)(m0 + row) * K + k0 + k4, &sA[row * 64 + k4]);
        } else {
            for (int idx = t; idx < 1024; idx += 256) {
                int row = idx >> 6, kk = idx & 63;
                int gk = k0 + kk;
                int cg = (gk < K) ? gk : 0;                    // clamp, no branch
                unsigned short v = load_a_elem(A + (long)(m0 + row) * K + cg);
                sA[row * 64 + kk] = (gk < K) ? v : (unsigned short)0;
            }
        }

        // ================= stage B tile [64 x 128] -> sB[n][k] =================
        if constexpr (!TRANS_B) {
            if (kFull && nFull) {
                const int n4 = (t & 31) * 4;            // 4 n-columns
                const int kb = (t >> 5) * 8;            // 8 k-rows
                fl4 r[8];
                #pragma unroll
                for (int j = 0; j < 8; ++j)
                    r[j] = *(const fl4*)(B + (long)(k0 + kb + j) * N + n0 + n4);
                #pragma unroll
                for (int nn = 0; nn < 4; ++nn) {
                    unsigned* d = (unsigned*)&sB[(n4 + nn) * 64 + kb];
                    d[0] = pack2_bf16(r[0][nn], r[1][nn]);
                    d[1] = pack2_bf16(r[2][nn], r[3][nn]);
                    d[2] = pack2_bf16(r[4][nn], r[5][nn]);
                    d[3] = pack2_bf16(r[6][nn], r[7][nn]);  // -> ds_store b128
                }
            } else {
                for (int idx = t; idx < 8192; idx += 256) {
                    int kk = idx >> 7, col = idx & 127;
                    int gk = k0 + kk, gn = n0 + col;
                    bool ok = (gk < K) && (gn < N);
                    long off = ok ? ((long)gk * N + gn) : 0;   // clamp
                    unsigned short v = f32_to_bf16(B[off]);
                    sB[col * 64 + kk] = ok ? v : (unsigned short)0;
                }
            }
        } else {
            if (kFull && nFull) {
                const int n  = t >> 1;                  // 0..127
                const int kb = (t & 1) * 32;            // 32 contiguous k
                fl4 r[8];
                #pragma unroll
                for (int j = 0; j < 8; ++j)
                    r[j] = *(const fl4*)(B + (long)(n0 + n) * K + k0 + kb + j * 4);
                unsigned* d = (unsigned*)&sB[n * 64 + kb];
                #pragma unroll
                for (int j = 0; j < 8; ++j) {
                    d[2 * j]     = pack2_bf16(r[j][0], r[j][1]);
                    d[2 * j + 1] = pack2_bf16(r[j][2], r[j][3]);
                }
            } else {
                for (int idx = t; idx < 8192; idx += 256) {
                    int kk = idx >> 7, col = idx & 127;
                    int gk = k0 + kk, gn = n0 + col;
                    bool ok = (gk < K) && (gn < N);
                    long off = ok ? ((long)gn * K + gk) : 0;   // clamp
                    unsigned short v = f32_to_bf16(B[off]);
                    sB[col * 64 + kk] = ok ? v : (unsigned short)0;
                }
            }
        }

        // ---- prefetch next K-chunk of B (global_prefetch_b8) ----
        if (k0 + 64 < K && nFull) {
            if constexpr (!TRANS_B)
                __builtin_prefetch(B + (long)(k0 + 64 + (t >> 5)) * N + n0 + (t & 31) * 4, 0, 1);
            else
                __builtin_prefetch(B + (long)(n0 + (t >> 1)) * K + k0 + 64 + (t & 1) * 32, 0, 1);
        }
        __syncthreads();

        // ================= 2x WMMA per stage =================
        // A 16x32 layout: lane(0-15)=M row, elem i -> K=i+16*half
        // B 32x16 layout: lane&15=N col,    elem i -> K=i+16*half
        const unsigned* sA32 = (const unsigned*)sA;   // [16][32] dwords
        const unsigned* sB32 = (const unsigned*)sB;   // [128][32] dwords
        #pragma unroll
        for (int j = 0; j < 2; ++j) {
            FragBF16 fa, fb;
            #pragma unroll
            for (int p = 0; p < 8; ++p) {
                int ka = 2 * p + 16 * half;  // even
                fa.u[p] = sA32[lr * 32 + j * 16 + (ka >> 1)];
                fb.u[p] = sB32[(wave * 16 + lr) * 32 + j * 16 + p + 8 * half];
            }
            acc = __builtin_amdgcn_wmma_f32_16x16x32_bf16(
                      false, fa.v, false, fb.v, (short)0, acc, false, false);
        }
        __syncthreads();
    }

    // ---- epilogue: D layout -> M = r + 8*half, N = lane&15 ----
    const int gn = n0 + wave * 16 + lr;
    if (gn < N) {
        const float bv = bias ? bias[gn] : 0.0f;
        #pragma unroll
        for (int r = 0; r < 8; ++r) {
            int gm = m0 + r + 8 * half;
            float v = acc[r] + bv;
            if (ACT == 1) v = (v > 0.0f) ? v : 0.2f * v;
            if constexpr (sizeof(OT) == 2) C[(long)gm * N + gn] = f32_to_bf16(v);
            else                           C[(long)gm * N + gn] = v;
        }
    }
}

// adj[i,j] = D[i] * A[j,i] * D[j],  D = rowsum(A)^-0.5   (80x80)
__global__ void genadj_kernel(const float* __restrict__ A, float* __restrict__ adj) {
    __shared__ float D[80];
    const int t = threadIdx.x;  // 128
    if (t < 80) {
        float s = 0.0f;
        for (int j = 0; j < 80; ++j) s += A[t * 80 + j];
        D[t] = rsqrtf(s);
    }
    __syncthreads();
    for (int idx = t; idx < 6400; idx += 128) {
        int i = idx / 80, j = idx - i * 80;
        adj[idx] = D[i] * A[j * 80 + i] * D[j];
    }
}

// bias2[c] = sum_j b_img[j] * cls[c,j]   (cls bf16 [80,8000])
__global__ void bias2_kernel(const float* __restrict__ b_img,
                             const unsigned short* __restrict__ cls,
                             float* __restrict__ bias2) {
    __shared__ float red[256];
    const int c = blockIdx.x, t = threadIdx.x;
    float s = 0.0f;
    for (int j = t; j < 8000; j += 256)
        s += b_img[j] * bf16_to_f32(cls[c * 8000 + j]);
    red[t] = s;
    __syncthreads();
    for (int w = 128; w > 0; w >>= 1) {
        if (t < w) red[t] += red[t + w];
        __syncthreads();
    }
    if (t == 0) bias2[c] = red[0];
}

extern "C" void kernel_launch(void* const* d_in, const int* in_sizes, int n_in,
                              void* d_out, int out_size, void* d_ws, size_t ws_size,
                              hipStream_t stream) {
    (void)in_sizes; (void)n_in; (void)out_size; (void)ws_size;

    const float* feature = (const float*)d_in[0];  // [1024,2048]
    const float* inp     = (const float*)d_in[1];  // [1,80,300]
    const float* A       = (const float*)d_in[2];  // [80,80]
    const float* W_gc1   = (const float*)d_in[3];  // [300,1024]
    const float* W_gc2   = (const float*)d_in[4];  // [1024,2048]
    const float* W_img   = (const float*)d_in[5];  // [8000,2048]
    const float* b_img   = (const float*)d_in[6];  // [8000]
    const float* W_cls   = (const float*)d_in[7];  // [8000,2048]
    const float* b_cls   = (const float*)d_in[8];  // [8000]
    float*       out     = (float*)d_out;          // [1024,80]

    // workspace carve-up (~3.9 MB total)
    float* ws    = (float*)d_ws;
    float* adj   = ws;                 // 80*80
    float* t1    = adj   + 6400;       // 80*1024
    float* x1    = t1    + 81920;      // 80*1024
    float* t2    = x1    + 81920;      // 80*2048
    float* x2    = t2    + 163840;     // 80*2048
    float* WmixT = x2    + 163840;     // 80*2048 = (cls @ W_img)
    float* bias2 = WmixT + 163840;     // 80
    unsigned short* cls = (unsigned short*)(bias2 + 96);  // [80,8000] bf16

    // 1) adj = gen_adj(A)
    genadj_kernel<<<1, 128, 0, stream>>>(A, adj);

    // 2) t1 = inp0 @ W_gc1                          [80,300]x[300,1024]
    gemm_wmma_bf16<float, false, 0, float>
        <<<dim3(8, 5), 256, 0, stream>>>(inp, W_gc1, nullptr, t1, 80, 1024, 300);

    // 3) x1 = leaky_relu(adj @ t1, 0.2)             [80,80]x[80,1024]
    gemm_wmma_bf16<float, false, 1, float>
        <<<dim3(8, 5), 256, 0, stream>>>(adj, t1, nullptr, x1, 80, 1024, 80);

    // 4) t2 = x1 @ W_gc2                            [80,1024]x[1024,2048]
    gemm_wmma_bf16<float, false, 0, float>
        <<<dim3(16, 5), 256, 0, stream>>>(x1, W_gc2, nullptr, t2, 80, 2048, 1024);

    // 5) x2 = adj @ t2                              [80,80]x[80,2048]
    gemm_wmma_bf16<float, false, 0, float>
        <<<dim3(16, 5), 256, 0, stream>>>(adj, t2, nullptr, x2, 80, 2048, 80);

    // 6) cls = bf16(x2 @ W_cls^T + b_cls)           [80,2048]x[2048,8000]
    gemm_wmma_bf16<float, true, 0, unsigned short>
        <<<dim3(63, 5), 256, 0, stream>>>(x2, W_cls, b_cls, cls, 80, 8000, 2048);

    // 7) WmixT = cls @ W_img                        [80,8000]x[8000,2048]
    gemm_wmma_bf16<unsigned short, false, 0, float>
        <<<dim3(16, 5), 256, 0, stream>>>(cls, W_img, nullptr, WmixT, 80, 2048, 8000);

    // 8) bias2[c] = b_img . cls[c,:]
    bias2_kernel<<<80, 256, 0, stream>>>(b_img, cls, bias2);

    // 9) out = feature @ WmixT^T + bias2            [1024,2048]x[2048,80]
    gemm_wmma_bf16<float, true, 0, float>
        <<<dim3(1, 64), 256, 0, stream>>>(feature, WmixT, bias2, out, 1024, 80, 2048);
}
